// NestedGIN_eff_18932215841157
// MI455X (gfx1250) — matrix operand order repaired and checked by reading, not compiled, not run on hardware
//
#include <hip/hip_runtime.h>
#include <hip/hip_bf16.h>
#include <math.h>

// ---------------------------------------------------------------------------
// NestedGIN on MI455X (gfx1250): wave32 + v_wmma_f32_16x16x32_f16 for all
// 64-wide GEMMs; edge features stored f16; sorted segments done without
// atomics; unsorted dst scatter via global_atomic_add_f32.
// Round 2: __launch_bounds__(256,1) to kill scratch spills in the cached-B
// WMMA kernels; BN0 scale/shift staged in LDS for the z-MLP A-loader.
// ---------------------------------------------------------------------------

typedef _Float16 h16;
typedef __attribute__((ext_vector_type(16))) _Float16 v16h;
typedef __attribute__((ext_vector_type(8)))  float    v8f;

#define DEV static __device__ __forceinline__

static constexpr int Nn  = 100000;
static constexpr int Ee  = 1000000;
static constexpr int Pp  = 2000000;
static constexpr int Gg  = 512;
static constexpr int HCAT = 320;   // 5 * 64

DEV v8f wmma16(v16h a, v16h b, v8f c) {
  // 8 args: (neg_a, A, neg_b, B, c_mod, C, reuse_a, reuse_b)
  return __builtin_amdgcn_wmma_f32_16x16x32_f16(false, a, false, b, (short)0, c,
                                                false, false);
}

// B tile loader: W row-major [K x ldn]; B-operand 32x16 layout:
// lanes 0-15: col n, K = kb+0..7 and kb+16..23; lanes 16-31: K = +8 offset.
DEV v16h load_b_w(const float* W, int ldn, int nbase, int kb,
                  int kmax, int nmax, int lane) {
  int n  = nbase + (lane & 15);
  int k0 = kb + ((lane & 16) ? 8 : 0);
  v16h b;
#pragma unroll
  for (int i = 0; i < 16; ++i) {
    int k = k0 + (i < 8 ? i : i + 8);
    float w = (k < kmax && n < nmax) ? W[(long)k * ldn + n] : 0.0f;
    b[i] = (h16)w;
  }
  return b;
}

// A tile loader from f16 matrix, row-major [M x ld]. A 16x32 layout:
// lanes 0-15: row r, K = kb+0..7 and kb+16..23; lanes 16-31: K = +8 offset.
DEV v16h load_a_h(const h16* A, int ld, int Mbase, int kb, int lane) {
  const h16* p = A + (long)(Mbase + (lane & 15)) * ld;
  int ko = kb + ((lane & 16) ? 8 : 0);
  v16h a;
#pragma unroll
  for (int i = 0; i < 8; ++i) { a[i] = p[ko + i]; a[i + 8] = p[ko + 16 + i]; }
  return a;
}

// A tile loader from f32 matrix (convert to f16 on the fly).
DEV v16h load_a_f32(const float* A, int ld, int Mbase, int kb, int lane) {
  const float* p = A + (long)(Mbase + (lane & 15)) * ld;
  int ko = kb + ((lane & 16) ? 8 : 0);
  v16h a;
#pragma unroll
  for (int i = 0; i < 8; ++i) {
    a[i]     = (h16)p[ko + i];
    a[i + 8] = (h16)p[ko + 16 + i];
  }
  return a;
}

// ---------------------------------------------------------------------------
// K1: z_raw[e, 0:64] = sum_{p in segment e} pos_enc[p] * z_init[pos_index[p]]
// pos_batch sorted -> one wave per edge, binary-search segment, no atomics.
// ---------------------------------------------------------------------------
__global__ __launch_bounds__(256, 1)
void k_z_accum(const int* __restrict__ pos_index,
               const float* __restrict__ pos_enc,
               const int* __restrict__ pos_batch,
               const float* __restrict__ z_init,
               float* __restrict__ z_raw) {
  int gtid = blockIdx.x * blockDim.x + threadIdx.x;
  int wave = gtid >> 5, lane = gtid & 31;
  if (wave >= Ee) return;
  int lo = 0, hi = Pp;
  while (lo < hi) { int m = (lo + hi) >> 1; if (pos_batch[m] <  wave) lo = m + 1; else hi = m; }
  int s = lo; hi = Pp;
  while (lo < hi) { int m = (lo + hi) >> 1; if (pos_batch[m] <= wave) lo = m + 1; else hi = m; }
  int e = lo;
  float a0 = 0.f, a1 = 0.f;
  for (int p = s; p < e; ++p) {
    float w = pos_enc[p];
    const float* zr = z_init + (long)pos_index[p] * 64;
    a0 += w * zr[lane];
    a1 += w * zr[lane + 32];
  }
  float* o = z_raw + (long)wave * 64;
  o[lane] = a0; o[lane + 32] = a1;
}

// ---------------------------------------------------------------------------
// K2: zf16 = relu(bn1(lin(relu(bn0(z_raw)))))   [E x 64] -> f16, WMMA GEMM.
// BN0 scale/shift staged in LDS (read via ds_load in the streaming loop).
// ---------------------------------------------------------------------------
__global__ __launch_bounds__(256, 1)
void k_z_mlp(const float* __restrict__ zr,
             const float* s0sc, const float* s0bi, const float* s0me, const float* s0va,
             const float* __restrict__ W, const float* __restrict__ Wb,
             const float* s1sc, const float* s1bi, const float* s1me, const float* s1va,
             h16* __restrict__ zf, int tiles) {
  __shared__ float s0l[64], t0l[64];
  int tid = threadIdx.x;
  if (tid < 64) {
    float s = s0sc[tid] * rsqrtf(s0va[tid] + 1e-5f);
    s0l[tid] = s;
    t0l[tid] = s0bi[tid] - s0me[tid] * s;
  }
  __syncthreads();

  int gtid = blockIdx.x * blockDim.x + threadIdx.x;
  int wave = gtid >> 5, lane = gtid & 31;
  int nwaves = (gridDim.x * blockDim.x) >> 5;

  v16h B[2][4];
#pragma unroll
  for (int ks = 0; ks < 2; ++ks)
#pragma unroll
    for (int nt = 0; nt < 4; ++nt)
      B[ks][nt] = load_b_w(W, 64, nt * 16, ks * 32, 64, 64, lane);

  int n0 = lane & 15;
  float s1[4], t1[4], bb[4];
#pragma unroll
  for (int nt = 0; nt < 4; ++nt) {
    int c = nt * 16 + n0;
    float s = s1sc[c] * rsqrtf(s1va[c] + 1e-5f);
    s1[nt] = s; t1[nt] = s1bi[c] - s1me[c] * s; bb[nt] = Wb[c];
  }
  int mo = (lane & 16) ? 8 : 0;

  for (int t = wave; t < tiles; t += nwaves) {
    int Mb = t * 16;
    if (t + nwaves < tiles)
      __builtin_prefetch(zr + (long)(t + nwaves) * 16 * 64, 0, 1);
    v8f acc[4] = {{}, {}, {}, {}};
#pragma unroll
    for (int ks = 0; ks < 2; ++ks) {
      // A with fused bn0 + relu, f32 -> f16 (bn params from LDS)
      const float* p = zr + (long)(Mb + n0) * 64;
      int kbase = ks * 32 + mo;
      v16h a;
#pragma unroll
      for (int i = 0; i < 8; ++i) {
        int k = kbase + i;
        a[i] = (h16)fmaxf(fmaf(p[k], s0l[k], t0l[k]), 0.f);
        k += 16;
        a[i + 8] = (h16)fmaxf(fmaf(p[k], s0l[k], t0l[k]), 0.f);
      }
#pragma unroll
      for (int nt = 0; nt < 4; ++nt) acc[nt] = wmma16(a, B[ks][nt], acc[nt]);
    }
#pragma unroll
    for (int nt = 0; nt < 4; ++nt) {
      int c = nt * 16 + n0;
#pragma unroll
      for (int v = 0; v < 8; ++v) {
        int row = Mb + v + mo;
        float val = fmaxf((acc[nt][v] + bb[nt]) * s1[nt] + t1[nt], 0.f);
        zf[(long)row * 64 + c] = (h16)val;
      }
    }
  }
}

// ---------------------------------------------------------------------------
// K3: GINE message: msg = relu(x[src] + z @ We + be); atomic scatter to agg.
// ---------------------------------------------------------------------------
template <int NT>
__global__ __launch_bounds__(256, 1)
void k_msg(const h16* __restrict__ zf,
           const float* __restrict__ We, const float* __restrict__ be,
           const int* __restrict__ src, const int* __restrict__ dst,
           const float* __restrict__ x32, int ldx,
           const h16* __restrict__ x16,
           float* __restrict__ agg, int lda, int nout, int tiles) {
  int gtid = blockIdx.x * blockDim.x + threadIdx.x;
  int wave = gtid >> 5, lane = gtid & 31;
  int nwaves = (gridDim.x * blockDim.x) >> 5;

  v16h B[2][NT];
#pragma unroll
  for (int ks = 0; ks < 2; ++ks)
#pragma unroll
    for (int nt = 0; nt < NT; ++nt)
      B[ks][nt] = load_b_w(We, nout, nt * 16, ks * 32, 64, nout, lane);

  int n0 = lane & 15;
  float eb[NT];
#pragma unroll
  for (int nt = 0; nt < NT; ++nt) {
    int c = nt * 16 + n0;
    eb[nt] = (c < nout) ? be[c] : 0.f;
  }
  int mo = (lane & 16) ? 8 : 0;

  for (int t = wave; t < tiles; t += nwaves) {
    int Mb = t * 16;
    if (t + nwaves < tiles)
      __builtin_prefetch(zf + (long)(t + nwaves) * 16 * 64, 0, 1);
    v8f acc[NT];
#pragma unroll
    for (int nt = 0; nt < NT; ++nt) acc[nt] = v8f{};
#pragma unroll
    for (int ks = 0; ks < 2; ++ks) {
      v16h a = load_a_h(zf, 64, Mb, ks * 32, lane);
#pragma unroll
      for (int nt = 0; nt < NT; ++nt) acc[nt] = wmma16(a, B[ks][nt], acc[nt]);
    }
#pragma unroll
    for (int v = 0; v < 8; ++v) {
      int e  = Mb + v + mo;
      int se = src[e], de = dst[e];
#pragma unroll
      for (int nt = 0; nt < NT; ++nt) {
        int c = nt * 16 + n0;
        if (c < nout) {
          float xs = x32 ? x32[(long)se * ldx + c] : (float)x16[(long)se * 64 + c];
          float msg = fmaxf(acc[nt][v] + eb[nt] + xs, 0.f);
          atomicAdd(&agg[(long)de * lda + c], msg);
        }
      }
    }
  }
}

// ---------------------------------------------------------------------------
// K4a: t = relu(bn1( (alpha*in1 + in2) @ W + b ))  -> tmp f16 [M x 64]
// ---------------------------------------------------------------------------
template <int KS>
__global__ __launch_bounds__(256, 1)
void k_node_a(const float* __restrict__ in1, int ld1,
              const float* __restrict__ eps,
              const float* __restrict__ in2, int ld2, int K1,
              const float* __restrict__ W, const float* __restrict__ Wb,
              const float* bsc, const float* bbi, const float* bme, const float* bva,
              h16* __restrict__ outt, int tiles) {
  int gtid = blockIdx.x * blockDim.x + threadIdx.x;
  int wave = gtid >> 5, lane = gtid & 31;
  int nwaves = (gridDim.x * blockDim.x) >> 5;
  float alpha = 1.f + (eps ? eps[0] : 0.f);

  v16h B[KS][4];
#pragma unroll
  for (int ks = 0; ks < KS; ++ks)
#pragma unroll
    for (int nt = 0; nt < 4; ++nt)
      B[ks][nt] = load_b_w(W, 64, nt * 16, ks * 32, K1, 64, lane);

  int n0 = lane & 15;
  float s1[4], t1[4], bb[4];
#pragma unroll
  for (int nt = 0; nt < 4; ++nt) {
    int c = nt * 16 + n0;
    float s = bsc[c] * rsqrtf(bva[c] + 1e-5f);
    s1[nt] = s; t1[nt] = bbi[c] - bme[c] * s; bb[nt] = Wb[c];
  }
  int mo = (lane & 16) ? 8 : 0;

  for (int t = wave; t < tiles; t += nwaves) {
    int Mb = t * 16;
    v8f acc[4] = {{}, {}, {}, {}};
#pragma unroll
    for (int ks = 0; ks < KS; ++ks) {
      int r = Mb + n0;
      int kbase = ks * 32 + mo;
      v16h a;
#pragma unroll
      for (int i = 0; i < 8; ++i) {
        int k = kbase + i;
        float v0 = 0.f;
        if (k < K1) {
          v0 = alpha * in1[(long)r * ld1 + k];
          if (in2) v0 += in2[(long)r * ld2 + k];
        }
        a[i] = (h16)v0;
        k += 16;
        float v1 = 0.f;
        if (k < K1) {
          v1 = alpha * in1[(long)r * ld1 + k];
          if (in2) v1 += in2[(long)r * ld2 + k];
        }
        a[i + 8] = (h16)v1;
      }
#pragma unroll
      for (int nt = 0; nt < 4; ++nt) acc[nt] = wmma16(a, B[ks][nt], acc[nt]);
    }
#pragma unroll
    for (int nt = 0; nt < 4; ++nt) {
      int c = nt * 16 + n0;
#pragma unroll
      for (int v = 0; v < 8; ++v) {
        int row = Mb + v + mo;
        float val = fmaxf((acc[nt][v] + bb[nt]) * s1[nt] + t1[nt], 0.f);
        outt[(long)row * 64 + c] = (h16)val;
      }
    }
  }
}

// ---------------------------------------------------------------------------
// K4b: h = relu(bn2( t @ W + b )); write f32 (hcat block, ld 320) + f16 copy.
// ---------------------------------------------------------------------------
__global__ __launch_bounds__(256, 1)
void k_node_b(const h16* __restrict__ tin,
              const float* __restrict__ W, const float* __restrict__ Wb,
              const float* bsc, const float* bbi, const float* bme, const float* bva,
              float* __restrict__ out32, int ld32,
              h16* __restrict__ out16, int tiles) {
  int gtid = blockIdx.x * blockDim.x + threadIdx.x;
  int wave = gtid >> 5, lane = gtid & 31;
  int nwaves = (gridDim.x * blockDim.x) >> 5;

  v16h B[2][4];
#pragma unroll
  for (int ks = 0; ks < 2; ++ks)
#pragma unroll
    for (int nt = 0; nt < 4; ++nt)
      B[ks][nt] = load_b_w(W, 64, nt * 16, ks * 32, 64, 64, lane);

  int n0 = lane & 15;
  float s1[4], t1[4], bb[4];
#pragma unroll
  for (int nt = 0; nt < 4; ++nt) {
    int c = nt * 16 + n0;
    float s = bsc[c] * rsqrtf(bva[c] + 1e-5f);
    s1[nt] = s; t1[nt] = bbi[c] - bme[c] * s; bb[nt] = Wb[c];
  }
  int mo = (lane & 16) ? 8 : 0;

  for (int t = wave; t < tiles; t += nwaves) {
    int Mb = t * 16;
    v8f acc[4] = {{}, {}, {}, {}};
#pragma unroll
    for (int ks = 0; ks < 2; ++ks) {
      v16h a = load_a_h(tin, 64, Mb, ks * 32, lane);
#pragma unroll
      for (int nt = 0; nt < 4; ++nt) acc[nt] = wmma16(a, B[ks][nt], acc[nt]);
    }
#pragma unroll
    for (int nt = 0; nt < 4; ++nt) {
      int c = nt * 16 + n0;
#pragma unroll
      for (int v = 0; v < 8; ++v) {
        int row = Mb + v + mo;
        float val = fmaxf((acc[nt][v] + bb[nt]) * s1[nt] + t1[nt], 0.f);
        out32[(long)row * ld32 + c] = val;
        if (out16) out16[(long)row * 64 + c] = (h16)val;
      }
    }
  }
}

// ---------------------------------------------------------------------------
// K5: global mean pool per graph (batch sorted) -> pooled [G x 320].
// ---------------------------------------------------------------------------
__global__ __launch_bounds__(320, 1)
void k_pool(const float* __restrict__ hcat,
            const int* __restrict__ batch,
            float* __restrict__ pooled) {
  int g = blockIdx.x;
  int c = threadIdx.x;           // 320 threads
  int lo = 0, hi = Nn;
  while (lo < hi) { int m = (lo + hi) >> 1; if (batch[m] <  g) lo = m + 1; else hi = m; }
  int s = lo; hi = Nn;
  while (lo < hi) { int m = (lo + hi) >> 1; if (batch[m] <= g) lo = m + 1; else hi = m; }
  int e = lo;
  float acc = 0.f;
  for (int r = s; r < e; ++r) acc += hcat[(long)r * HCAT + c];
  pooled[(long)g * HCAT + c] = acc / fmaxf((float)(e - s), 1.f);
}

// ---------------------------------------------------------------------------
// K6: ph = relu(bn( pooled @ W(320x64) + b ))   M=512, K=320 (10 K-steps).
// ---------------------------------------------------------------------------
__global__ __launch_bounds__(256, 1)
void k_lin1(const float* __restrict__ A,
            const float* __restrict__ W, const float* __restrict__ Wb,
            const float* bsc, const float* bbi, const float* bme, const float* bva,
            float* __restrict__ out) {
  int gtid = blockIdx.x * blockDim.x + threadIdx.x;
  int wave = gtid >> 5, lane = gtid & 31;
  int nwaves = (gridDim.x * blockDim.x) >> 5;
  int n0 = lane & 15;
  float s1[4], t1[4], bb[4];
#pragma unroll
  for (int nt = 0; nt < 4; ++nt) {
    int c = nt * 16 + n0;
    float s = bsc[c] * rsqrtf(bva[c] + 1e-5f);
    s1[nt] = s; t1[nt] = bbi[c] - bme[c] * s; bb[nt] = Wb[c];
  }
  int mo = (lane & 16) ? 8 : 0;
  int tiles = Gg / 16;

  for (int t = wave; t < tiles; t += nwaves) {
    int Mb = t * 16;
    v8f acc[4] = {{}, {}, {}, {}};
    for (int kb = 0; kb < 10; ++kb) {
      v16h a = load_a_f32(A, HCAT, Mb, kb * 32, lane);
#pragma unroll
      for (int nt = 0; nt < 4; ++nt) {
        v16h b = load_b_w(W, 64, nt * 16, kb * 32, HCAT, 64, lane);
        acc[nt] = wmma16(a, b, acc[nt]);
      }
    }
#pragma unroll
    for (int nt = 0; nt < 4; ++nt) {
      int c = nt * 16 + n0;
#pragma unroll
      for (int v = 0; v < 8; ++v) {
        int row = Mb + v + mo;
        float val = fmaxf((acc[nt][v] + bb[nt]) * s1[nt] + t1[nt], 0.f);
        out[(long)row * 64 + c] = val;
      }
    }
  }
}

// ---------------------------------------------------------------------------
// K7: logits = ph @ W2(64x10) + b2; log_softmax. One thread per graph.
// ---------------------------------------------------------------------------
__global__ __launch_bounds__(256, 1)
void k_final(const float* __restrict__ ph,
             const float* __restrict__ W, const float* __restrict__ Wb,
             float* __restrict__ out) {
  int r = blockIdx.x * blockDim.x + threadIdx.x;
  if (r >= Gg) return;
  float lg[10];
#pragma unroll
  for (int c = 0; c < 10; ++c) {
    float a = Wb[c];
    for (int k = 0; k < 64; ++k) a += ph[(long)r * 64 + k] * W[k * 10 + c];
    lg[c] = a;
  }
  float m = lg[0];
#pragma unroll
  for (int c = 1; c < 10; ++c) m = fmaxf(m, lg[c]);
  float se = 0.f;
#pragma unroll
  for (int c = 0; c < 10; ++c) se += expf(lg[c] - m);
  float lse = logf(se);
#pragma unroll
  for (int c = 0; c < 10; ++c) out[(long)r * 10 + c] = lg[c] - m - lse;
}

// ---------------------------------------------------------------------------
extern "C" void kernel_launch(void* const* d_in, const int* in_sizes, int n_in,
                              void* d_out, int out_size, void* d_ws, size_t ws_size,
                              hipStream_t stream) {
  (void)in_sizes; (void)n_in; (void)out_size; (void)ws_size;

  const float* x      = (const float*)d_in[0];
  const int*   ei     = (const int*)d_in[1];
  const int*   batch  = (const int*)d_in[2];
  const int*   posidx = (const int*)d_in[3];
  const float* posenc = (const float*)d_in[4];
  const int*   posb   = (const int*)d_in[5];
  const int* src = ei;
  const int* dst = ei + Ee;

  // params flattened as a JAX pytree: dict keys alphabetical, lists in order.
  int ii = 6;
  auto F = [&]() { return (const float*)d_in[ii++]; };

  // bn_lin1: bias, mean, scale, var
  const float *bnl1_bi = F(), *bnl1_me = F(), *bnl1_sc = F(), *bnl1_va = F();
  // conv1: edge_lin{b,w}, eps, nn{bn1{b,m,s,v}, bn2{b,m,s,v}, l1{b,w}, l2{b,w}}
  const float *c1_eb = F(), *c1_ew = F(), *c1_eps = F();
  const float *c1_b1bi = F(), *c1_b1me = F(), *c1_b1sc = F(), *c1_b1va = F();
  const float *c1_b2bi = F(), *c1_b2me = F(), *c1_b2sc = F(), *c1_b2va = F();
  const float *c1_l1b = F(), *c1_l1w = F(), *c1_l2b = F(), *c1_l2w = F();
  // convs[3]: same structure as conv1 (15 leaves each)
  struct CV {
    const float *eb, *ew, *eps;
    const float *b1bi, *b1me, *b1sc, *b1va;
    const float *b2bi, *b2me, *b2sc, *b2va;
    const float *l1b, *l1w, *l2b, *l2w;
  } cv[3];
  for (int k = 0; k < 3; ++k) {
    cv[k].eb = F(); cv[k].ew = F(); cv[k].eps = F();
    cv[k].b1bi = F(); cv[k].b1me = F(); cv[k].b1sc = F(); cv[k].b1va = F();
    cv[k].b2bi = F(); cv[k].b2me = F(); cv[k].b2sc = F(); cv[k].b2va = F();
    cv[k].l1b = F(); cv[k].l1w = F(); cv[k].l2b = F(); cv[k].l2w = F();
  }
  // lin1{b,w}, lin2{b,w}
  const float *l1b = F(), *l1w = F(), *l2b = F(), *l2w = F();
  // x_emb: bn1, bn2, l1{b,w}, l2{b,w}
  const float *xe_b1bi = F(), *xe_b1me = F(), *xe_b1sc = F(), *xe_b1va = F();
  const float *xe_b2bi = F(), *xe_b2me = F(), *xe_b2sc = F(), *xe_b2va = F();
  const float *xe_l1b = F(), *xe_l1w = F(), *xe_l2b = F(), *xe_l2w = F();
  // z_bn0, z_bn1, z_init, z_lin{b,w}
  const float *zb0_bi = F(), *zb0_me = F(), *zb0_sc = F(), *zb0_va = F();
  const float *zb1_bi = F(), *zb1_me = F(), *zb1_sc = F(), *zb1_va = F();
  const float *z_init = F();
  const float *zl_b = F(), *zl_w = F();

  // ---- workspace carve ----
  auto au = [](size_t v) { return (v + 255) & ~(size_t)255; };
  char* p = (char*)d_ws;
  float* z_raw  = (float*)p; p += au((size_t)Ee * 64 * 4);   // 256 MB
  h16*   zf16   = (h16*)p;   p += au((size_t)Ee * 64 * 2);   // 128 MB
  float* agg    = (float*)p; p += au((size_t)Nn * 64 * 4);   // 25.6 MB
  h16*   tmp    = (h16*)p;   p += au((size_t)Nn * 64 * 2);   // 12.8 MB
  h16*   hcur   = (h16*)p;   p += au((size_t)Nn * 64 * 2);   // 12.8 MB
  float* hcat   = (float*)p; p += au((size_t)Nn * HCAT * 4); // 128 MB
  float* pooled = (float*)p; p += au((size_t)Gg * HCAT * 4);
  float* ph     = (float*)p; p += au((size_t)Gg * 64 * 4);

  const int etiles = Ee / 16;   // 62500
  const int ntiles = Nn / 16;   // 6250

  // 1) z accumulate (sorted segments, deterministic)
  k_z_accum<<<(Ee * 32) / 256, 256, 0, stream>>>(posidx, posenc, posb, z_init, z_raw);

  // 2) z MLP -> zf16  (WMMA)
  k_z_mlp<<<1024, 256, 0, stream>>>(z_raw,
                                    zb0_sc, zb0_bi, zb0_me, zb0_va,
                                    zl_w, zl_b,
                                    zb1_sc, zb1_bi, zb1_me, zb1_va,
                                    zf16, etiles);

  // 3) conv1: messages over edges (out dim 10, agg stride 16)
  hipMemsetAsync(agg, 0, (size_t)Nn * 16 * 4, stream);
  k_msg<1><<<1024, 256, 0, stream>>>(zf16, c1_ew, c1_eb, src, dst,
                                     x, 10, (const h16*)nullptr,
                                     agg, 16, 10, etiles);
  k_node_a<1><<<512, 256, 0, stream>>>(x, 10, c1_eps, agg, 16, 10,
                                       c1_l1w, c1_l1b,
                                       c1_b1sc, c1_b1bi, c1_b1me, c1_b1va,
                                       tmp, ntiles);
  k_node_b<<<512, 256, 0, stream>>>(tmp, c1_l2w, c1_l2b,
                                    c1_b2sc, c1_b2bi, c1_b2me, c1_b2va,
                                    hcat + 64, HCAT, hcur, ntiles);

  // 4) x_emb MLP -> hcat block 0
  k_node_a<1><<<512, 256, 0, stream>>>(x, 10, nullptr, nullptr, 0, 10,
                                       xe_l1w, xe_l1b,
                                       xe_b1sc, xe_b1bi, xe_b1me, xe_b1va,
                                       tmp, ntiles);
  k_node_b<<<512, 256, 0, stream>>>(tmp, xe_l2w, xe_l2b,
                                    xe_b2sc, xe_b2bi, xe_b2me, xe_b2va,
                                    hcat + 0, HCAT, (h16*)nullptr, ntiles);

  // 5) remaining conv layers
  for (int k = 0; k < 3; ++k) {
    hipMemsetAsync(agg, 0, (size_t)Nn * 64 * 4, stream);
    k_msg<4><<<1024, 256, 0, stream>>>(zf16, cv[k].ew, cv[k].eb, src, dst,
                                       (const float*)nullptr, 0, hcur,
                                       agg, 64, 64, etiles);
    k_node_a<2><<<512, 256, 0, stream>>>(hcat + 64 * (1 + k), HCAT, cv[k].eps,
                                         agg, 64, 64,
                                         cv[k].l1w, cv[k].l1b,
                                         cv[k].b1sc, cv[k].b1bi, cv[k].b1me, cv[k].b1va,
                                         tmp, ntiles);
    k_node_b<<<512, 256, 0, stream>>>(tmp, cv[k].l2w, cv[k].l2b,
                                      cv[k].b2sc, cv[k].b2bi, cv[k].b2me, cv[k].b2va,
                                      hcat + 64 * (2 + k), HCAT, hcur, ntiles);
  }

  // 6) pool, lin1 (WMMA K=320), head
  k_pool<<<Gg, HCAT, 0, stream>>>(hcat, batch, pooled);
  k_lin1<<<4, 256, 0, stream>>>(pooled, l1w, l1b,
                                bnl1_sc, bnl1_bi, bnl1_me, bnl1_va, ph);
  k_final<<<2, 256, 0, stream>>>(ph, l2w, l2b, (float*)d_out);
}
